// LSTMBlock_41953240547630
// MI455X (gfx1250) — matrix operand order repaired and checked by reading, not compiled, not run on hardware
//
#include <hip/hip_runtime.h>
#include <hip/hip_bf16.h>
#include <stdint.h>

#define T_  1024
#define B_  32
#define D_  512
#define H_  512
#define G_  2048                 // 4*H
#define M_  (T_ * B_)            // 32768 rows
#define KT_ (D_ / 32)            // 16 k-tiles of 32
#define NT_DIR (G_ / 16)         // 128 n-tiles per direction

#if __has_builtin(__builtin_amdgcn_sched_barrier)
#define SCHED_FENCE() __builtin_amdgcn_sched_barrier(0)
#else
#define SCHED_FENCE()
#endif

typedef __attribute__((ext_vector_type(16))) __bf16 bf16x16;
typedef __attribute__((ext_vector_type(8)))  float  f32x8;

union FragU { uint4 q[2]; bf16x16 v; };
union V8U   { f32x8 v; float4 q[2]; float f[8]; };

__device__ __forceinline__ unsigned short f2bf(float x) {
  unsigned u = __float_as_uint(x);
  unsigned r = u + 0x7fffu + ((u >> 16) & 1u);   // round-to-nearest-even
  return (unsigned short)(r >> 16);
}
__device__ __forceinline__ float sigmoidf_(float x) {
  return 1.0f / (1.0f + __expf(-x));
}
__device__ __forceinline__ float tanhf_(float x) {
  // tanh(x) = 2*sigmoid(2x) - 1, on the hardware exp pipe
  return 2.0f / (1.0f + __expf(-2.0f * x)) - 1.0f;
}
__device__ __forceinline__ f32x8 wmma_bf16(const bf16x16& a, const bf16x16& b, const f32x8& c) {
  return __builtin_amdgcn_wmma_f32_16x16x32_bf16(false, a, false, b, (short)0, c, false, false);
}

// ---------------- kernel 1a: f32 -> bf16 repack (weights) ----------------
__global__ __launch_bounds__(256) void k_f32_to_bf16(const float* __restrict__ s,
                                                     unsigned short* __restrict__ d, int n) {
  int i = (blockIdx.x * 256 + threadIdx.x) * 8;
  if (i >= n) return;
  float4 a = *(const float4*)(s + i);
  float4 b = *(const float4*)(s + i + 4);
  uint4 o;
  o.x = (unsigned)f2bf(a.x) | ((unsigned)f2bf(a.y) << 16);
  o.y = (unsigned)f2bf(a.z) | ((unsigned)f2bf(a.w) << 16);
  o.z = (unsigned)f2bf(b.x) | ((unsigned)f2bf(b.y) << 16);
  o.w = (unsigned)f2bf(b.z) | ((unsigned)f2bf(b.w) << 16);
  *(uint4*)(d + i) = o;
}

// ---------------- kernel 1b: LayerNorm + bf16 cast, one wave per row ------
__global__ __launch_bounds__(256) void k_layernorm_bf16(const float* __restrict__ x,
                                                        const float* __restrict__ gamma,
                                                        const float* __restrict__ beta,
                                                        unsigned short* __restrict__ xnbf) {
  int lane = threadIdx.x & 31, wid = threadIdx.x >> 5;
  int row = blockIdx.x * 8 + wid;
  const float* xr = x + (size_t)row * D_;
  float v[16], s = 0.f, ss = 0.f;
#pragma unroll
  for (int i = 0; i < 4; ++i) {
    float4 t = *(const float4*)(xr + lane * 16 + i * 4);
    v[4*i+0] = t.x; v[4*i+1] = t.y; v[4*i+2] = t.z; v[4*i+3] = t.w;
    s += t.x + t.y + t.z + t.w;
    ss += t.x*t.x + t.y*t.y + t.z*t.z + t.w*t.w;
  }
#pragma unroll
  for (int m = 16; m >= 1; m >>= 1) {
    s  += __shfl_xor(s,  m, 32);
    ss += __shfl_xor(ss, m, 32);
  }
  float mu = s * (1.0f / D_);
  float var = ss * (1.0f / D_) - mu * mu;
  float rs = rsqrtf(var + 1e-5f);
  unsigned short o[16];
#pragma unroll
  for (int i = 0; i < 16; ++i) {
    int col = lane * 16 + i;
    o[i] = f2bf((v[i] - mu) * rs * gamma[col] + beta[col]);
  }
  uint4 p0, p1;
  p0.x = o[0] | (o[1] << 16);  p0.y = o[2]  | (o[3]  << 16);
  p0.z = o[4] | (o[5] << 16);  p0.w = o[6]  | (o[7]  << 16);
  p1.x = o[8] | (o[9] << 16);  p1.y = o[10] | (o[11] << 16);
  p1.z = o[12]| (o[13]<< 16);  p1.w = o[14] | (o[15] << 16);
  unsigned short* dst = xnbf + (size_t)row * D_ + lane * 16;
  *(uint4*)dst = p0;
  *(uint4*)(dst + 8) = p1;
}

// ---------------- kernel 2: gates GEMM  gx = xn @ W_ih^T + b_ih + b_hh ----
// Output tile-swizzled: gxfrag[dir][m_tile*128 + n_tile][lane*8 + e]  (C layout)
// Software-pipelined; sched_barrier pins stage-(k+1) loads above stage-k WMMAs.
__global__ __launch_bounds__(256) void k_gates_gemm(const unsigned short* __restrict__ xnbf,
                                                    const unsigned short* __restrict__ wih_bf,
                                                    const float* __restrict__ bih_f,
                                                    const float* __restrict__ bhh_f,
                                                    const float* __restrict__ bih_b,
                                                    const float* __restrict__ bhh_b,
                                                    float* __restrict__ gxfrag) {
  int lane = threadIdx.x & 31, wid = threadIdx.x >> 5;
  long w = (long)blockIdx.x * 8 + wid;            // 131072 waves
  int ngrp = (int)(w & 63);                       // 64 groups of 4 n-tiles (both dirs)
  int mt   = (int)(w >> 6);                       // 0..2047
  int dir  = ngrp >> 5;
  int ntd0 = (ngrp & 31) * 4;                     // first n-tile within dir
  int ln   = lane & 15;
  const float* bih = dir ? bih_b : bih_f;
  const float* bhh = dir ? bhh_b : bhh_f;

  V8U acc[4];
#pragma unroll
  for (int gi = 0; gi < 4; ++gi) {
    int nl = (ntd0 + gi) * 16 + ln;
    float bias = bih[nl] + bhh[nl];
#pragma unroll
    for (int e = 0; e < 8; ++e) acc[gi].f[e] = bias;
  }

  const unsigned short* arow = xnbf + (size_t)(mt * 16 + ln) * D_;
  const unsigned short* wbase = wih_bf + (size_t)dir * G_ * D_;
  const unsigned short* brow[4];
#pragma unroll
  for (int gi = 0; gi < 4; ++gi)
    brow[gi] = wbase + (size_t)((ntd0 + gi) * 16 + ln) * D_;
  int alo = (lane < 16) ? 0 : 8;
  int blo = (lane < 16) ? 0 : 8;

  FragU A[2], Bf[2][4];
  // stage 0 loads (kt = 0)
  A[0].q[0] = *(const uint4*)(arow + alo);
  A[0].q[1] = *(const uint4*)(arow + alo + 16);
#pragma unroll
  for (int gi = 0; gi < 4; ++gi) {
    const unsigned short* bp = brow[gi] + blo;
    Bf[0][gi].q[0] = *(const uint4*)bp;
    Bf[0][gi].q[1] = *(const uint4*)(bp + 16);
  }
  SCHED_FENCE();

#pragma unroll
  for (int kt = 0; kt < KT_; ++kt) {
    const int cur = kt & 1, nxt = cur ^ 1;
    if (kt + 1 < KT_) {                           // issue next-stage loads first
      int kb2 = (kt + 1) * 32;
      A[nxt].q[0] = *(const uint4*)(arow + kb2 + alo);
      A[nxt].q[1] = *(const uint4*)(arow + kb2 + alo + 16);
#pragma unroll
      for (int gi = 0; gi < 4; ++gi) {
        const unsigned short* bp = brow[gi] + kb2 + blo;
        Bf[nxt][gi].q[0] = *(const uint4*)bp;
        Bf[nxt][gi].q[1] = *(const uint4*)(bp + 16);
      }
    }
    SCHED_FENCE();                                // loads(kt+1) stay above wmma(kt)
#pragma unroll
    for (int gi = 0; gi < 4; ++gi)
      acc[gi].v = wmma_bf16(A[cur].v, Bf[cur][gi].v, acc[gi].v);
  }

  float* gdst = gxfrag + (size_t)dir * M_ * G_;
#pragma unroll
  for (int gi = 0; gi < 4; ++gi) {
    long tile = (long)mt * NT_DIR + (ntd0 + gi);
    float* p = gdst + tile * 256 + lane * 8;
    *(float4*)p       = acc[gi].q[0];
    *(float4*)(p + 4) = acc[gi].q[1];
  }
}

// ---------------- kernel 3: persistent LSTM scan (1 WG per direction) ----
__global__ __launch_bounds__(1024, 1) void k_lstm_scan(const float* __restrict__ gxfrag,
                                                       const unsigned short* __restrict__ whh_bf,
                                                       float* __restrict__ hbuf) {
  __shared__ unsigned short h_lds[B_ * H_];       // 32 KB, h in bf16
  int dir = blockIdx.x;
  int lane = threadIdx.x & 31, wid = threadIdx.x >> 5;   // 32 waves
  int ln = lane & 15;
  int jb = wid * 16;                              // hidden slice owned by wave

  for (int i = threadIdx.x; i < B_ * H_ / 2; i += 1024) ((unsigned*)h_lds)[i] = 0u;
  __syncthreads();

  const float* gxd = gxfrag + (size_t)dir * M_ * G_;
  const unsigned short* wd = whh_bf + (size_t)dir * G_ * D_;
  float* hd = hbuf + (size_t)dir * M_ * H_;

  const unsigned short* brow[4];
#pragma unroll
  for (int g = 0; g < 4; ++g)
    brow[g] = wd + (size_t)(g * 512 + jb + ln) * H_;
  int alo = (lane < 16) ? 0 : 8;
  int blo = (lane < 16) ? 0 : 8;

  V8U c[2];
#pragma unroll
  for (int m = 0; m < 2; ++m)
#pragma unroll
    for (int e = 0; e < 8; ++e) c[m].f[e] = 0.f;

#pragma unroll 1
  for (int t = 0; t < T_; ++t) {
    int ct = dir ? (T_ - 1 - t) : t;

    V8U acc[4][2];                                // acc init = gx tile (C-fragment order)
#pragma unroll
    for (int g = 0; g < 4; ++g)
#pragma unroll
      for (int m = 0; m < 2; ++m) {
        long tile = (long)(ct * 2 + m) * NT_DIR + (g * 32 + wid);
        const float* p = gxd + tile * 256 + lane * 8;
        acc[g][m].q[0] = *(const float4*)p;
        acc[g][m].q[1] = *(const float4*)(p + 4);
      }

    // prefetch next timestep's gx tiles (global_prefetch_b8) while we compute
    if (t + 1 < T_) {
      int nt = dir ? (T_ - 2 - t) : (t + 1);
#pragma unroll
      for (int g = 0; g < 4; ++g)
#pragma unroll
        for (int m = 0; m < 2; ++m) {
          long tile = (long)(nt * 2 + m) * NT_DIR + (g * 32 + wid);
          __builtin_prefetch(gxd + tile * 256 + lane * 8, 0, 3);
        }
    }

    // gates += h @ W_hh^T, double-buffered weight fragments
    FragU Bf[2][4];
#pragma unroll
    for (int g = 0; g < 4; ++g) {
      const unsigned short* bp = brow[g] + blo;
      Bf[0][g].q[0] = *(const uint4*)bp;
      Bf[0][g].q[1] = *(const uint4*)(bp + 16);
    }
    SCHED_FENCE();
#pragma unroll
    for (int kt = 0; kt < KT_; ++kt) {
      const int cur = kt & 1, nxt = cur ^ 1;
      if (kt + 1 < KT_) {
        int kb2 = (kt + 1) * 32;
#pragma unroll
        for (int g = 0; g < 4; ++g) {
          const unsigned short* bp = brow[g] + kb2 + blo;
          Bf[nxt][g].q[0] = *(const uint4*)bp;
          Bf[nxt][g].q[1] = *(const uint4*)(bp + 16);
        }
      }
      int kb = kt * 32;
      FragU A0, A1;
      const unsigned short* a0 = h_lds + (size_t)ln * H_ + kb + alo;
      const unsigned short* a1 = h_lds + (size_t)(16 + ln) * H_ + kb + alo;
      A0.q[0] = *(const uint4*)a0; A0.q[1] = *(const uint4*)(a0 + 16);
      A1.q[0] = *(const uint4*)a1; A1.q[1] = *(const uint4*)(a1 + 16);
      SCHED_FENCE();                              // loads(kt+1) stay above wmma(kt)
#pragma unroll
      for (int g = 0; g < 4; ++g) {
        acc[g][0].v = wmma_bf16(A0.v, Bf[cur][g].v, acc[g][0].v);
        acc[g][1].v = wmma_bf16(A1.v, Bf[cur][g].v, acc[g][1].v);
      }
    }

    float hf[2][8]; unsigned short hb[2][8];
#pragma unroll
    for (int m = 0; m < 2; ++m)
#pragma unroll
      for (int e = 0; e < 8; ++e) {
        float iv = sigmoidf_(acc[0][m].f[e]);
        float fv = sigmoidf_(acc[1][m].f[e]);
        float gv = tanhf_(acc[2][m].f[e]);
        float ov = sigmoidf_(acc[3][m].f[e]);
        float cv = fv * c[m].f[e] + iv * gv;
        c[m].f[e] = cv;
        float hv = ov * tanhf_(cv);
        hf[m][e] = hv; hb[m][e] = f2bf(hv);
      }

    __syncthreads();                              // all waves done reading old h
#pragma unroll
    for (int m = 0; m < 2; ++m)
#pragma unroll
      for (int e = 0; e < 8; ++e) {
        int b = m * 16 + e + ((lane >= 16) ? 8 : 0);   // C layout: M = e (+8 hi half)
        int j = jb + ln;
        h_lds[b * H_ + j] = hb[m][e];
        hd[(size_t)(ct * B_ + b) * H_ + j] = hf[m][e];
      }
    __syncthreads();                              // new h visible to everyone
  }
}

// ---------------- kernel 4: out = h_f + h_b ------------------------------
__global__ __launch_bounds__(256) void k_sum_out(const float* __restrict__ a,
                                                 const float* __restrict__ b,
                                                 float* __restrict__ o, int n4) {
  int i = blockIdx.x * 256 + threadIdx.x;
  if (i >= n4) return;
  float4 x = ((const float4*)a)[i];
  float4 y = ((const float4*)b)[i];
  float4 r; r.x = x.x + y.x; r.y = x.y + y.y; r.z = x.z + y.z; r.w = x.w + y.w;
  ((float4*)o)[i] = r;
}

extern "C" void kernel_launch(void* const* d_in, const int* in_sizes, int n_in,
                              void* d_out, int out_size, void* d_ws, size_t ws_size,
                              hipStream_t stream) {
  const float* x      = (const float*)d_in[0];
  const float* gamma  = (const float*)d_in[1];
  const float* beta   = (const float*)d_in[2];
  const float* w_ih_f = (const float*)d_in[3];
  const float* w_hh_f = (const float*)d_in[4];
  const float* b_ih_f = (const float*)d_in[5];
  const float* b_hh_f = (const float*)d_in[6];
  const float* w_ih_b = (const float*)d_in[7];
  const float* w_hh_b = (const float*)d_in[8];
  const float* b_ih_b = (const float*)d_in[9];
  const float* b_hh_b = (const float*)d_in[10];
  float* out = (float*)d_out;

  char* ws = (char*)d_ws;
  size_t off = 0;
  unsigned short* xnbf  = (unsigned short*)(ws + off); off += (size_t)M_ * D_ * 2;
  unsigned short* wihbf = (unsigned short*)(ws + off); off += (size_t)2 * G_ * D_ * 2;
  unsigned short* whhbf = (unsigned short*)(ws + off); off += (size_t)2 * G_ * D_ * 2;
  float* gxfrag         = (float*)(ws + off);          off += (size_t)2 * M_ * G_ * 4;
  float* hbuf           = (float*)(ws + off);          off += (size_t)2 * M_ * H_ * 4;

  const int WN = G_ * D_;                 // 1,048,576 elems per weight matrix
  const int cvt_blocks = WN / (256 * 8);  // 512
  k_f32_to_bf16<<<cvt_blocks, 256, 0, stream>>>(w_ih_f, wihbf, WN);
  k_f32_to_bf16<<<cvt_blocks, 256, 0, stream>>>(w_ih_b, wihbf + (size_t)G_ * D_, WN);
  k_f32_to_bf16<<<cvt_blocks, 256, 0, stream>>>(w_hh_f, whhbf, WN);
  k_f32_to_bf16<<<cvt_blocks, 256, 0, stream>>>(w_hh_b, whhbf + (size_t)G_ * D_, WN);

  k_layernorm_bf16<<<M_ / 8, 256, 0, stream>>>(x, gamma, beta, xnbf);

  // 2048 m-tiles * 64 n-groups = 131072 waves / 8 per block
  k_gates_gemm<<<16384, 256, 0, stream>>>(xnbf, wihbf, b_ih_f, b_hh_f, b_ih_b, b_hh_b, gxfrag);

  k_lstm_scan<<<2, 1024, 0, stream>>>(gxfrag, whhbf, hbuf);

  int n4 = M_ * H_ / 4;                   // 4,194,304
  k_sum_out<<<n4 / 256, 256, 0, stream>>>(hbuf, hbuf + (size_t)M_ * H_, out, n4);
}
